// ScaledDotProductAttention_63634235458009
// MI455X (gfx1250) — matrix operand order repaired and checked by reading, not compiled, not run on hardware
//
#include <hip/hip_runtime.h>
#include <math.h>

// ---------------------------------------------------------------------------
// CDNA5 (gfx1250, wave32) scaled dot-product attention.
//   Phase 1: Q/K/V = x @ W + b   via v_wmma_f32_16x16x32_bf16  (K stored [s][d],
//            V stored transposed [d][s] for the attn@V B-operand).
//   Phase 2: flash attention, 32-key blocks, online softmax, WMMA for both
//            Q*K^T and P*V. Scores never touch HBM.
// Workspace: Q(8MB) + K(8MB) + Vt(8MB) bf16 = 24MB.
// ---------------------------------------------------------------------------

typedef __attribute__((ext_vector_type(16))) __bf16 v16bf;
typedef __attribute__((ext_vector_type(8)))  float  v8f;

struct __attribute__((aligned(16))) U128 { unsigned int x, y, z, w; };
union ABPack { v16bf v; U128 q[2]; };

static __device__ __forceinline__ v8f wmma_bf16(v16bf a, v16bf b, v8f c) {
  // D = A(16x32 bf16) * B(32x16 bf16) + C(16x16 f32)
  return __builtin_amdgcn_wmma_f32_16x16x32_bf16(
      /*neg_a=*/false, a, /*neg_b=*/false, b,
      /*c_mod=*/(short)0, c, /*reuse_a=*/false, /*reuse_b=*/false);
}

// ===========================================================================
// Phase 1: QKV projection.  One wave -> one 16x16 output tile; 8 waves/block
// share one W column-tile (staged transposed bf16 in LDS).
// grid = 3 mats * 128 row-blocks * 16 col-tiles = 6144 blocks of 256 threads.
// ===========================================================================
__global__ __launch_bounds__(256) void qkv_proj_kernel(
    const float* __restrict__ x,
    const float* __restrict__ Wq, const float* __restrict__ bq,
    const float* __restrict__ Wk, const float* __restrict__ bk,
    const float* __restrict__ Wv, const float* __restrict__ bv,
    __bf16* __restrict__ qout, __bf16* __restrict__ kout,
    __bf16* __restrict__ vtout)
{
  __shared__ __align__(16) __bf16 lds_wt[16][256];   // W^T tile: [n][k], 8KB

  const int bid      = blockIdx.x;
  const int mat      = bid / 2048;          // 0=Q 1=K 2=V
  const int rem      = bid % 2048;
  const int col_tile = rem & 15;
  const int row_blk  = rem >> 4;            // 0..127 (128 rows each)
  const int tid      = threadIdx.x;
  const int wave     = tid >> 5;
  const int lane     = tid & 31;
  const int hf       = lane >> 4;           // wave32 half
  const int lm       = lane & 15;
  const int col0     = col_tile * 16;

  const float* W    = (mat == 0) ? Wq : (mat == 1) ? Wk : Wv;
  const float* bias = (mat == 0) ? bq : (mat == 1) ? bk : bv;

  // Cooperative load + transpose of W[:, col0:col0+16] -> lds_wt[n][k] (bf16).
  {
    const int k = tid;                                  // 0..255
    const float* wrow = W + (size_t)k * 256 + col0;     // 16 contiguous floats
    float4 w0 = *(const float4*)(wrow + 0);
    float4 w1 = *(const float4*)(wrow + 4);
    float4 w2 = *(const float4*)(wrow + 8);
    float4 w3 = *(const float4*)(wrow + 12);
    float wf[16] = {w0.x, w0.y, w0.z, w0.w, w1.x, w1.y, w1.z, w1.w,
                    w2.x, w2.y, w2.z, w2.w, w3.x, w3.y, w3.z, w3.w};
#pragma unroll
    for (int n = 0; n < 16; ++n) lds_wt[n][k] = (__bf16)wf[n];
  }
  __syncthreads();

  const int row0 = (row_blk * 8 + wave) * 16;
  const int m    = row0 + lm;                           // this lane's A-row
  const float* xrow = x + (size_t)m * 256;

  v8f acc = {};
#pragma unroll
  for (int c = 0; c < 8; ++c) {                         // K = 256 in 8 x 32
    // A operand: x rows, f32 -> bf16.  e<8: k=c*32+hf*8+e ; e>=8: +16.
    const float* p0 = xrow + c * 32 + hf * 8;
    float4 f0 = ((const float4*)p0)[0];
    float4 f1 = ((const float4*)p0)[1];
    float4 f2 = ((const float4*)(p0 + 16))[0];
    float4 f3 = ((const float4*)(p0 + 16))[1];
    v16bf av;
    av[0]=(__bf16)f0.x; av[1]=(__bf16)f0.y; av[2]=(__bf16)f0.z; av[3]=(__bf16)f0.w;
    av[4]=(__bf16)f1.x; av[5]=(__bf16)f1.y; av[6]=(__bf16)f1.z; av[7]=(__bf16)f1.w;
    av[8]=(__bf16)f2.x; av[9]=(__bf16)f2.y; av[10]=(__bf16)f2.z; av[11]=(__bf16)f2.w;
    av[12]=(__bf16)f3.x; av[13]=(__bf16)f3.y; av[14]=(__bf16)f3.z; av[15]=(__bf16)f3.w;
    // B operand: lane holds column n=lm, e<8: k=c*32+hf*8+e ; e>=8: +16.
    ABPack bp;
    const U128* kp = (const U128*)&lds_wt[lm][c * 32 + hf * 8];
    bp.q[0] = kp[0]; bp.q[1] = kp[2];
    acc = wmma_bf16(av, bp.v, acc);
  }

  const float bcol = bias[col0 + lm];
#pragma unroll
  for (int j = 0; j < 8; ++j) {                         // C/D: M = j + 8*hf
    const int   rg  = row0 + j + 8 * hf;
    const int   col = col0 + lm;
    const __bf16 v  = (__bf16)(acc[j] + bcol);
    if (mat == 0) {
      qout[(size_t)rg * 256 + col] = v;
    } else if (mat == 1) {
      kout[(size_t)rg * 256 + col] = v;
    } else {                                            // V stored transposed
      const int bb = rg >> 11, s = rg & 2047;
      vtout[((size_t)(bb * 256 + col) << 11) + s] = v;
    }
  }
}

// ===========================================================================
// Phase 2: flash attention.  4 waves/block, 16 queries/wave.
// grid = 8 batches * 32 query-blocks = 256 blocks of 128 threads.
// LDS: K-tile 16KB + Vt-tile 16KB + P staging 4KB = 36KB.
// ===========================================================================
__global__ __launch_bounds__(128) void flash_attn_kernel(
    const __bf16* __restrict__ Q, const __bf16* __restrict__ K,
    const __bf16* __restrict__ Vt, float* __restrict__ out)
{
  __shared__ __align__(16) __bf16 lds_K[32][256];      // [key][d]
  __shared__ __align__(16) __bf16 lds_V[256][32];      // [d][key]  (V^T)
  __shared__ __align__(16) __bf16 lds_P[4][16][32];    // per-wave P restripe

  const int b    = blockIdx.x >> 5;
  const int qblk = blockIdx.x & 31;
  const int tid  = threadIdx.x;
  const int wave = tid >> 5;
  const int lane = tid & 31;
  const int hf   = lane >> 4;
  const int lm   = lane & 15;
  const int q0   = qblk * 64 + wave * 16;

  // Preload this wave's Q tile as 8 A-operand chunks (reused for all 64 k-blocks).
  v16bf qa[8];
  {
    const __bf16* qrow = Q + ((size_t)(b * 2048 + q0 + lm)) * 256;
#pragma unroll
    for (int c = 0; c < 8; ++c) {
      ABPack a;
      a.q[0] = *(const U128*)(qrow + c * 32 + hf * 8);
      a.q[1] = *(const U128*)(qrow + c * 32 + 16 + hf * 8);
      qa[c] = a.v;
    }
  }

  v8f o[16];                                           // O acc: 16 d-tiles
#pragma unroll
  for (int ct = 0; ct < 16; ++ct) o[ct] = (v8f){};
  float mrow[8], lrow[8];
#pragma unroll
  for (int j = 0; j < 8; ++j) { mrow[j] = -3.0e38f; lrow[j] = 0.0f; }

  const __bf16* kbase = K  + (size_t)b * 2048 * 256;
  const __bf16* vbase = Vt + (size_t)b * 256 * 2048;
  const float scale = 0.0625f;                         // 1/sqrt(256)

  for (int kb = 0; kb < 64; ++kb) {
    const int k0 = kb * 32;
    // ---- cooperative tile loads -----------------------------------------
    {
      const int r  = tid >> 2;                         // key row 0..31
      const int d0 = (tid & 3) * 64;
      const U128* src = (const U128*)(kbase + (size_t)(k0 + r) * 256 + d0);
      U128*       dst = (U128*)&lds_K[r][d0];
#pragma unroll
      for (int i = 0; i < 8; ++i) dst[i] = src[i];
    }
    {
      const int d = tid * 2;                           // two d-rows of V^T
#pragma unroll
      for (int rr = 0; rr < 2; ++rr) {
        const U128* src = (const U128*)(vbase + (size_t)(d + rr) * 2048 + k0);
        U128*       dst = (U128*)&lds_V[d + rr][0];
#pragma unroll
        for (int i = 0; i < 4; ++i) dst[i] = src[i];
      }
    }
    if (kb < 63) {                                     // gfx1250 global_prefetch_b8
      __builtin_prefetch(kbase + (size_t)(k0 + 32 + (tid >> 2)) * 256, 0, 3);
      __builtin_prefetch(vbase + (size_t)(tid * 2) * 2048 + k0 + 32, 0, 3);
    }
    __syncthreads();

    // ---- S = Q * K^T : two 16x16 tiles, 8 accumulating WMMAs each --------
    v8f s0 = (v8f){}, s1 = (v8f){};
#pragma unroll
    for (int c = 0; c < 8; ++c) {
      ABPack b0, b1;
      const U128* kp0 = (const U128*)&lds_K[lm][c * 32 + hf * 8];
      const U128* kp1 = (const U128*)&lds_K[16 + lm][c * 32 + hf * 8];
      b0.q[0] = kp0[0]; b0.q[1] = kp0[2];
      b1.q[0] = kp1[0]; b1.q[1] = kp1[2];
      s0 = wmma_bf16(qa[c], b0.v, s0);
      s1 = wmma_bf16(qa[c], b1.v, s1);
    }

    // ---- online softmax (row M = j + 8*hf lives on lanes lm=0..15) -------
    float p0[8], p1[8];
#pragma unroll
    for (int j = 0; j < 8; ++j) {
      const float a0 = s0[j] * scale;
      const float a1 = s1[j] * scale;
      float rmax = fmaxf(a0, a1);
#pragma unroll
      for (int off = 8; off >= 1; off >>= 1)           // stays within half-wave
        rmax = fmaxf(rmax, __shfl_xor(rmax, off, 32));
      const float mnew  = fmaxf(mrow[j], rmax);
      const float alpha = __expf(mrow[j] - mnew);
      const float e0 = __expf(a0 - mnew);
      const float e1 = __expf(a1 - mnew);
      float rsum = e0 + e1;
#pragma unroll
      for (int off = 8; off >= 1; off >>= 1)
        rsum += __shfl_xor(rsum, off, 32);
      lrow[j] = lrow[j] * alpha + rsum;
      mrow[j] = mnew;
      p0[j] = e0; p1[j] = e1;
#pragma unroll
      for (int ct = 0; ct < 16; ++ct) o[ct][j] *= alpha;
    }

    // ---- restripe P: C-layout -> A-layout via per-wave LDS ---------------
#pragma unroll
    for (int j = 0; j < 8; ++j) {
      const int M = j + 8 * hf;
      lds_P[wave][M][lm]      = (__bf16)p0[j];
      lds_P[wave][M][16 + lm] = (__bf16)p1[j];
    }
    __threadfence_block();                             // LDS in-order per wave
    ABPack pa;
    pa.q[0] = *(const U128*)&lds_P[wave][lm][hf * 8];
    pa.q[1] = *(const U128*)&lds_P[wave][lm][16 + hf * 8];

    // ---- O += P * V : 16 accumulating WMMAs ------------------------------
#pragma unroll
    for (int ct = 0; ct < 16; ++ct) {
      ABPack vb;
      const U128* vp = (const U128*)&lds_V[ct * 16 + lm][hf * 8];
      vb.q[0] = vp[0]; vb.q[1] = vp[2];
      o[ct] = wmma_bf16(pa.v, vb.v, o[ct]);
    }
    __syncthreads();                                   // before tile reuse
  }

  // ---- epilogue: normalize and store f32 ---------------------------------
#pragma unroll
  for (int j = 0; j < 8; ++j) {
    const float inv = 1.0f / lrow[j];
    const int s = q0 + j + 8 * hf;
    float* orow = out + (size_t)(b * 2048 + s) * 256;
#pragma unroll
    for (int ct = 0; ct < 16; ++ct)
      orow[ct * 16 + lm] = o[ct][j] * inv;
  }
}

// ===========================================================================
extern "C" void kernel_launch(void* const* d_in, const int* in_sizes, int n_in,
                              void* d_out, int out_size, void* d_ws, size_t ws_size,
                              hipStream_t stream)
{
  const float* x  = (const float*)d_in[0];
  const float* Wq = (const float*)d_in[1];
  const float* bq = (const float*)d_in[2];
  const float* Wk = (const float*)d_in[3];
  const float* bk = (const float*)d_in[4];
  const float* Wv = (const float*)d_in[5];
  const float* bv = (const float*)d_in[6];
  float* out = (float*)d_out;

  const size_t ntok = (size_t)8 * 2048 * 256;          // 4M elements
  __bf16* qws = (__bf16*)d_ws;
  __bf16* kws = qws + ntok;
  __bf16* vws = kws + ntok;                            // total 24MB bf16

  qkv_proj_kernel<<<6144, 256, 0, stream>>>(x, Wq, bq, Wk, bk, Wv, bv,
                                            qws, kws, vws);
  flash_attn_kernel<<<256, 128, 0, stream>>>(qws, kws, vws, out);
}
